// WhiteNoiseConvolution2D_31241592111461
// MI455X (gfx1250) — matrix operand order repaired and checked by reading, compiled, not run on hardware
//
#include <hip/hip_runtime.h>
#include <hip/hip_bf16.h>

// ---------------------------------------------------------------------------
// Problem constants: x (32,128,128,64) f32 NHWC, kernel (3,3,64,128) f32 HWIO,
// out (32,128,128,128) f32.  Implicit GEMM: M=524288, N=128, K=576.
// ---------------------------------------------------------------------------
#define NIMG 32
#define HDIM 128
#define WDIM 128
#define CIN  64
#define COUT 128
#define ACT_ELEMS (NIMG * HDIM * WDIM * CIN)   // 33,554,432
#define WGT_ELEMS (3 * 3 * CIN * COUT)         // 73,728
#define KCHUNKS   18                           // 576 / 32
#define LDS_WSTR  72                           // halves; 144B row stride (16B-aligned)

typedef __attribute__((ext_vector_type(16))) __bf16 v16bf;
typedef __attribute__((ext_vector_type(8)))  __bf16 v8bf;
typedef __attribute__((ext_vector_type(8)))  float  v8f;

// ---------------------------------------------------------------------------
// Numerical-Recipes J1 (fp32), matches the jax reference formula.
// ---------------------------------------------------------------------------
__device__ __forceinline__ float j1f_nr(float x) {
    float ax = fabsf(x);
    float y = x * x;
    float p1 = x * (72362614232.0f + y * (-7895059235.0f + y * (242396853.1f
             + y * (-2972611.439f + y * (15704.48260f + y * (-30.16036606f))))));
    float q1 = 144725228442.0f + y * (2300535178.0f + y * (18583304.74f
             + y * (99447.43394f + y * (376.9991397f + y))));
    float small = p1 / q1;
    float axs = ax < 8.0f ? 8.0f : ax;
    float z = 8.0f / axs;
    float y2 = z * z;
    float xx = axs - 2.356194491f;
    float p = 1.0f + y2 * (0.183105e-2f + y2 * (-0.3516396496e-4f
            + y2 * (0.2457520174e-5f + y2 * (-0.240337019e-6f))));
    float q = 0.04687499995f + y2 * (-0.2002690873e-3f + y2 * (0.8449199096e-5f
            + y2 * (-0.88228987e-6f + y2 * 0.105787412e-6f)));
    float sgn = (x > 0.0f ? 1.0f : (x < 0.0f ? -1.0f : 0.0f));
    float big = sqrtf(0.636619772f / axs) * (cosf(xx) * p - z * sinf(xx) * q) * sgn;
    return ax < 8.0f ? small : big;
}

// ---------------------------------------------------------------------------
// Prep 1: act = bf16( j1(x) / (x+1) ), NHWC layout (same as x).
// ---------------------------------------------------------------------------
__global__ void act_prep_kernel(const float* __restrict__ x,
                                __bf16* __restrict__ act, int n) {
    int i = blockIdx.x * blockDim.x + threadIdx.x;
    if (i < n) {
        float v = x[i];
        float f = j1f_nr(v) / (v + 1.0f);
        act[i] = (__bf16)f;
    }
}

// ---------------------------------------------------------------------------
// Prep 2: rearrange HWIO weights into WMMA-B fragment order.
// wB[((t*2+g)*128 + n)*16 + e] = bf16( kernel[kk*128 + n] ),  kk = t*32+g*16+e.
// (B order used here, contiguous k: lane l -> column n=l%16, group g=l/16 holds K = g*16 + e.)
// ---------------------------------------------------------------------------
__global__ void weight_prep_kernel(const float* __restrict__ k,
                                   __bf16* __restrict__ wB) {
    int i = blockIdx.x * blockDim.x + threadIdx.x;   // 0 .. 73727
    if (i < WGT_ELEMS) {
        int e  = i & 15;
        int nc = (i >> 4) & 127;
        int tg = i >> 11;                 // t*2 + g, 0..35
        int kk = tg * 16 + e;             // = t*32 + g*16 + e
        wB[i] = (__bf16)k[kk * COUT + nc];
    }
}

// ---------------------------------------------------------------------------
// Conv: implicit GEMM with v_wmma_f32_16x16x32_bf16.
// Block = 256 threads (8 waves). Tile = 64 w-pixels x 128 channels.
// Waves: 4 (pixel) x 2 (channel); each wave: 16 pixels x 64 ch = 4 C tiles.
// ---------------------------------------------------------------------------
__global__ __launch_bounds__(256) void
conv_wmma_kernel(const __bf16* __restrict__ act,
                 const __bf16* __restrict__ wB,
                 float* __restrict__ out) {
    __shared__ __attribute__((aligned(16))) __bf16 sA[3 * 66 * LDS_WSTR];

    const int bid  = blockIdx.x;            // ((n*128 + h)*2 + wt)
    const int wt   = bid & 1;
    const int h    = (bid >> 1) & 127;
    const int nimg = bid >> 8;
    const int w0   = wt * 64;
    const int tid  = threadIdx.x;

    // ---- Stage input halo: rows h-1..h+1, w0-1..w0+64, 64 ch (bf16) --------
    const long actBase = (long)nimg * HDIM * WDIM * CIN;
    for (int idx = tid; idx < 198 * 8; idx += 256) {
        int pos = idx >> 3, seg = idx & 7;       // 8 x 16B segments per row pos
        int dh = pos / 66, wi = pos - dh * 66;
        int hh = h - 1 + dh;
        int ww = w0 - 1 + wi;
        uint4 v = make_uint4(0u, 0u, 0u, 0u);    // zero padding (f(0)=0)
        if ((unsigned)hh < (unsigned)HDIM && (unsigned)ww < (unsigned)WDIM) {
            v = *(const uint4*)(act + actBase + ((long)hh * WDIM + ww) * CIN + seg * 8);
        }
        *(uint4*)(&sA[(dh * 66 + wi) * LDS_WSTR + seg * 8]) = v;
    }
    __syncthreads();

    const int wid   = tid >> 5;
    const int lane  = tid & 31;
    const int waveM = wid >> 1;       // 0..3 : which 16-pixel row group
    const int waveN = wid & 1;        // 0..1 : which 64-channel group
    const int m     = lane & 15;      // A row / B,C,D column within tile
    const int g     = lane >> 4;      // lane group

    v8f c0 = {}, c1 = {}, c2 = {}, c3 = {};

    const int pixRow = waveM * 16 + m;                         // A-row pixel in tile
    const long bCol  = (long)(waveN * 64 + m);                 // B column base

#pragma unroll
    for (int t = 0; t < KCHUNKS; ++t) {
        const int pq   = t >> 1;              // 0..8 -> (dh,dw)
        const int half = t & 1;               // which 32-channel half
        const int dh   = pq / 3;
        const int dw   = pq - 3 * dh;

        // A fragment (contiguous k order, same as B): lane m, elems K = g*16+e
        const int rowOff = (dh * 66 + pixRow + dw) * LDS_WSTR + half * 32 + g * 16;
        v8bf alo = *(const v8bf*)(&sA[rowOff]);
        v8bf ahi = *(const v8bf*)(&sA[rowOff + 8]);
        v16bf a  = __builtin_shufflevector(alo, ahi,
                       0, 1, 2, 3, 4, 5, 6, 7, 8, 9, 10, 11, 12, 13, 14, 15);

        // B fragments: 16 contiguous bf16 per lane, coalesced across lanes.
        const __bf16* bp = wB + ((long)(t * 2 + g) * COUT + bCol) * 16;
        v8bf b0l = *(const v8bf*)(bp + 0 * 256);
        v8bf b0h = *(const v8bf*)(bp + 0 * 256 + 8);
        v8bf b1l = *(const v8bf*)(bp + 1 * 256);
        v8bf b1h = *(const v8bf*)(bp + 1 * 256 + 8);
        v8bf b2l = *(const v8bf*)(bp + 2 * 256);
        v8bf b2h = *(const v8bf*)(bp + 2 * 256 + 8);
        v8bf b3l = *(const v8bf*)(bp + 3 * 256);
        v8bf b3h = *(const v8bf*)(bp + 3 * 256 + 8);
        v16bf b0 = __builtin_shufflevector(b0l, b0h, 0,1,2,3,4,5,6,7,8,9,10,11,12,13,14,15);
        v16bf b1 = __builtin_shufflevector(b1l, b1h, 0,1,2,3,4,5,6,7,8,9,10,11,12,13,14,15);
        v16bf b2 = __builtin_shufflevector(b2l, b2h, 0,1,2,3,4,5,6,7,8,9,10,11,12,13,14,15);
        v16bf b3 = __builtin_shufflevector(b3l, b3h, 0,1,2,3,4,5,6,7,8,9,10,11,12,13,14,15);

        // D = A*B + C   (f32 accumulate)
        c0 = __builtin_amdgcn_wmma_f32_16x16x32_bf16(false, a, false, b0, (short)0, c0, false, false);
        c1 = __builtin_amdgcn_wmma_f32_16x16x32_bf16(false, a, false, b1, (short)0, c1, false, false);
        c2 = __builtin_amdgcn_wmma_f32_16x16x32_bf16(false, a, false, b2, (short)0, c2, false, false);
        c3 = __builtin_amdgcn_wmma_f32_16x16x32_bf16(false, a, false, b3, (short)0, c3, false, false);
    }

    // ---- Write out: D(r,lane) = (M = r + 8*g, N = lane%16) ------------------
    float* outRow = out + ((long)nimg * HDIM + h) * WDIM * COUT;
    const int chBase = waveN * 64 + m;
#pragma unroll
    for (int r = 0; r < 8; ++r) {
        const int w = w0 + waveM * 16 + r + 8 * g;
        float* o = outRow + (long)w * COUT + chBase;
        o[0]  = c0[r];
        o[16] = c1[r];
        o[32] = c2[r];
        o[48] = c3[r];
    }
}

// ---------------------------------------------------------------------------
// Harness entry
// ---------------------------------------------------------------------------
extern "C" void kernel_launch(void* const* d_in, const int* in_sizes, int n_in,
                              void* d_out, int out_size, void* d_ws, size_t ws_size,
                              hipStream_t stream) {
    const float* x = (const float*)d_in[0];     // 33,554,432 f32
    const float* k = (const float*)d_in[1];     // 73,728 f32
    float* out = (float*)d_out;                 // 67,108,864 f32

    __bf16* actBf = (__bf16*)d_ws;                                   // 67.1 MB
    __bf16* wB    = (__bf16*)((char*)d_ws + (size_t)ACT_ELEMS * 2);  // 147 KB

    act_prep_kernel<<<ACT_ELEMS / 256, 256, 0, stream>>>(x, actBf, ACT_ELEMS);
    weight_prep_kernel<<<WGT_ELEMS / 256, 256, 0, stream>>>(k, wB);

    // 32 images * 128 rows * 2 half-row tiles = 8192 blocks
    conv_wmma_kernel<<<NIMG * HDIM * 2, 256, 0, stream>>>(actBf, wB, out);
}
